// BondBreakGNN_17695265259649
// MI455X (gfx1250) — matrix operand order repaired and checked by reading, not compiled, not run on hardware
//
#include <hip/hip_runtime.h>
#include <hip/hip_bf16.h>

typedef __attribute__((ext_vector_type(16))) __bf16          v16bf;
typedef __attribute__((ext_vector_type(8)))  unsigned int    v8ui;
typedef __attribute__((ext_vector_type(8)))  float           v8f;

#define HIDF 64

// ---------- bf16 split helpers (x ~= hi + lo, each bf16, truncation split) ----------

__device__ __forceinline__ void split2(float v, unsigned short& h, unsigned short& l) {
  unsigned int xb = __builtin_bit_cast(unsigned int, v);
  unsigned int hb = xb & 0xffff0000u;
  h = (unsigned short)(hb >> 16);
  float rem = v - __builtin_bit_cast(float, hb);
  l = (unsigned short)(__builtin_bit_cast(unsigned int, rem) >> 16);
}

// A-fragment (16x32 bf16, ISA 7.12.2): lane m=lane&15, hb=lane>>4.
// elements 0..7 -> K = off0..off0+7 ; elements 8..15 -> K = off1..off1+7.
// Packed as 8 u32s; {bf16hi(v1),bf16hi(v0)} pack is a single v_perm_b32.
__device__ __forceinline__ void split_a(const float* __restrict__ row,
                                        int off0, int off1,
                                        v16bf& ah, v16bf& al) {
  v8ui H, L;
#pragma unroll
  for (int p = 0; p < 8; ++p) {
    const int idx = (p < 4) ? (off0 + 2 * p) : (off1 + 2 * (p - 4));
    const float v0 = row[idx], v1 = row[idx + 1];
    const unsigned int b0 = __builtin_bit_cast(unsigned int, v0);
    const unsigned int b1 = __builtin_bit_cast(unsigned int, v1);
    H[p] = __builtin_amdgcn_perm(b1, b0, 0x07060302u);   // {b1[31:16], b0[31:16]}
    const float r0 = v0 - __builtin_bit_cast(float, b0 & 0xffff0000u);
    const float r1 = v1 - __builtin_bit_cast(float, b1 & 0xffff0000u);
    L[p] = __builtin_amdgcn_perm(__builtin_bit_cast(unsigned int, r1),
                                 __builtin_bit_cast(unsigned int, r0), 0x07060302u);
  }
  ah = __builtin_bit_cast(v16bf, H);
  al = __builtin_bit_cast(v16bf, L);
}

// Stage W[K x 64] (K = nch*32) into LDS, pre-swizzled into WMMA B-fragment
// register layout: fragment q = c*4+nt, lane, elem i contiguous (32B per lane).
// B layout (ISA): lane n=lane&15, elem i -> K = c*32 + (i>=8?16:0) + (lane>=16?8:0) + (i&7).
__device__ __forceinline__ void stage_b(const float* __restrict__ W,
                                        unsigned short* __restrict__ s_bh,
                                        unsigned short* __restrict__ s_bl,
                                        int nch, int tid, int nthreads) {
  const int total = nch * 4 * 32 * 16;
  for (int f = tid; f < total; f += nthreads) {
    const int i    = f & 15;
    const int lane = (f >> 4) & 31;
    const int q    = f >> 9;
    const int nt   = q & 3;
    const int c    = q >> 2;
    const int K = c * 32 + ((i >> 3) << 4) + ((lane >> 4) << 3) + (i & 7);
    const int N = nt * 16 + (lane & 15);
    unsigned short h, l;
    split2(W[K * HIDF + N], h, l);
    s_bh[f] = h;
    s_bl[f] = l;
  }
}

__device__ __forceinline__ v16bf load_frag(const unsigned short* __restrict__ s,
                                           int q, int lane) {
  return *(const v16bf*)(s + (q * 32 + lane) * 16);      // 32B-aligned per lane
}

__device__ __forceinline__ v8f wmma_bf16(v16bf a, v16bf b, v8f c) {
  return __builtin_amdgcn_wmma_f32_16x16x32_bf16(false, a, false, b, (short)0, c,
                                                 false, false);
}

// One K-chunk of the split-precision GEMM, term-major so dependent WMMAs on the
// same accumulator are 4 instructions apart (hides WMMA->WMMA hazard slots).
__device__ __forceinline__ void mma_chunk(const unsigned short* __restrict__ s_bh,
                                          const unsigned short* __restrict__ s_bl,
                                          int c, int lane, v16bf ah, v16bf al,
                                          v8f acc[4]) {
  v16bf bh[4], bl[4];
#pragma unroll
  for (int nt = 0; nt < 4; ++nt) {
    bh[nt] = load_frag(s_bh, c * 4 + nt, lane);
    bl[nt] = load_frag(s_bl, c * 4 + nt, lane);
  }
#pragma unroll
  for (int nt = 0; nt < 4; ++nt) acc[nt] = wmma_bf16(ah, bh[nt], acc[nt]);
#pragma unroll
  for (int nt = 0; nt < 4; ++nt) acc[nt] = wmma_bf16(ah, bl[nt], acc[nt]);
#pragma unroll
  for (int nt = 0; nt < 4; ++nt) acc[nt] = wmma_bf16(al, bh[nt], acc[nt]);
}

// ---------- degree / normalization ----------

__global__ void k_deg_init(float* __restrict__ deg, int n) {
  int i = blockIdx.x * blockDim.x + threadIdx.x;
  if (i < n) deg[i] = 1.0f;   // self-loop
}

__global__ void k_deg_count(const int* __restrict__ dst, float* __restrict__ deg, int e) {
  int i = blockIdx.x * blockDim.x + threadIdx.x;
  if (i < e) atomicAdd(&deg[dst[i]], 1.0f);
}

__global__ void k_rsqrt(float* __restrict__ deg, int n) {
  int i = blockIdx.x * blockDim.x + threadIdx.x;
  if (i < n) deg[i] = rsqrtf(deg[i]);
}

// ---------- node GEMM: OUT[N x 64] = A[N x K] @ W[K x 64], bf16-split WMMA ----------

template <int KCH>   // K = KCH*32
__global__ void gemm_node(const float* __restrict__ A, const float* __restrict__ W,
                          float* __restrict__ OUT, int nrows) {
  __shared__ __align__(32) unsigned short s_bh[KCH * 2048];
  __shared__ __align__(32) unsigned short s_bl[KCH * 2048];
  stage_b(W, s_bh, s_bl, KCH, threadIdx.x, blockDim.x);
  __syncthreads();

  const int wave = threadIdx.x >> 5, lane = threadIdx.x & 31;
  const int tile = blockIdx.x * (blockDim.x >> 5) + wave;
  if (tile * 16 >= nrows) return;              // wave-uniform: EXEC stays all-ones
  const int m = lane & 15, hb = lane >> 4;
  const int K = KCH * 32;

  const float* arow = A + (size_t)(tile * 16 + m) * K;
  v8f acc[4] = {};
#pragma unroll
  for (int kc = 0; kc < KCH; ++kc) {
    const int kb = kc * 32;
    v16bf ah, al;
    split_a(arow, kb + hb * 8, kb + 16 + hb * 8, ah, al);
    mma_chunk(s_bh, s_bl, kc, lane, ah, al, acc);
  }
  float* ob = OUT + (size_t)tile * 16 * HIDF;
#pragma unroll
  for (int nt = 0; nt < 4; ++nt)
#pragma unroll
    for (int r = 0; r < 8; ++r)
      ob[(size_t)(r + hb * 8) * HIDF + nt * 16 + m] = acc[nt][r];
}

// ---------- GCN aggregation ----------

__global__ void k_agg_init(const float* __restrict__ h, const float* __restrict__ dinv,
                           float* __restrict__ agg, long long total) {
  long long g = (long long)blockIdx.x * blockDim.x + threadIdx.x;
  if (g >= total) return;
  int i = (int)(g >> 6);
  float di = dinv[i];
  agg[g] = di * di * h[g];                     // self-loop contribution
}

__global__ void k_agg_edge(const int* __restrict__ src, const int* __restrict__ dst,
                           const float* __restrict__ dinv, const float* __restrict__ h,
                           float* __restrict__ agg, long long total) {
  long long g = (long long)blockIdx.x * blockDim.x + threadIdx.x;
  if (g >= total) return;
  int e = (int)(g >> 6), f = (int)(g & 63);
  int s = src[e], d = dst[e];
  float c = dinv[s] * dinv[d];
  atomicAdd(&agg[(size_t)d * HIDF + f], c * h[(size_t)s * HIDF + f]);
}

__global__ void k_agg_fin(const float* __restrict__ agg, const float* __restrict__ b,
                          float* __restrict__ hout, long long total) {
  long long g = (long long)blockIdx.x * blockDim.x + threadIdx.x;
  if (g >= total) return;
  int f = (int)(g & 63);
  float v = agg[g] + b[f];
  hout[g] = v > 0.0f ? v : 0.0f;
}

// ---------- edge MLP: out[e] = relu([h2[s],h2[d],attr] @ W3 + b3) @ W4 + b4 ----------

__global__ void edge_mlp(const float* __restrict__ h2,
                         const int* __restrict__ src, const int* __restrict__ dst,
                         const float* __restrict__ attr,
                         const float* __restrict__ W3, const float* __restrict__ b3,
                         const float* __restrict__ W4, const float* __restrict__ b4,
                         float* __restrict__ out, int nedges) {
  __shared__ __align__(32) unsigned short s_bh[4 * 2048];   // W3 rows 0..127
  __shared__ __align__(32) unsigned short s_bl[4 * 2048];
  __shared__ float s_w3c[HIDF], s_b3[HIDF], s_w4[HIDF];
  __shared__ float s_z[8][16][HIDF + 1];        // padded: conflict-free reduction

  stage_b(W3, s_bh, s_bl, 4, threadIdx.x, blockDim.x);
  if (threadIdx.x < HIDF) {
    s_w3c[threadIdx.x] = W3[128 * HIDF + threadIdx.x];
    s_b3[threadIdx.x]  = b3[threadIdx.x];
    s_w4[threadIdx.x]  = W4[threadIdx.x];
  }
  __syncthreads();

  const int wave = threadIdx.x >> 5, lane = threadIdx.x & 31;
  const int tile = blockIdx.x * (blockDim.x >> 5) + wave;
  if (tile * 16 >= nedges) return;              // wave-uniform
  const int m = lane & 15, hb = lane >> 4;
  const int e0 = tile * 16;

  const int rows[2] = { src[e0 + m], dst[e0 + m] };
  v8f acc[4] = {};
#pragma unroll
  for (int half = 0; half < 2; ++half) {
    const float* arow = h2 + (size_t)rows[half] * HIDF;
#pragma unroll
    for (int kc = 0; kc < 2; ++kc) {
      const int kb = kc * 32;
      v16bf ah, al;
      split_a(arow, kb + hb * 8, kb + 16 + hb * 8, ah, al);
      mma_chunk(s_bh, s_bl, half * 2 + kc, lane, ah, al, acc);
    }
  }

  // epilogue: add attr rank-1 term + b3, ReLU, stash to LDS
  float att[8];
#pragma unroll
  for (int r = 0; r < 8; ++r) att[r] = attr[e0 + r + hb * 8];
#pragma unroll
  for (int nt = 0; nt < 4; ++nt) {
    const int j = nt * 16 + m;
    const float b3j = s_b3[j], w3c = s_w3c[j];
#pragma unroll
    for (int r = 0; r < 8; ++r) {
      float v = acc[nt][r] + b3j + att[r] * w3c;
      s_z[wave][r + hb * 8][j] = v > 0.0f ? v : 0.0f;
    }
  }
  // per-edge dot with W4 (same-wave LDS: DS ops stay in order per wave)
  if (lane < 16) {
    float s = 0.0f;
#pragma unroll 8
    for (int j = 0; j < HIDF; ++j) s += s_z[wave][lane][j] * s_w4[j];
    out[e0 + lane] = s + b4[0];
  }
}

// ---------- host-side orchestration ----------

extern "C" void kernel_launch(void* const* d_in, const int* in_sizes, int n_in,
                              void* d_out, int out_size, void* d_ws, size_t ws_size,
                              hipStream_t stream) {
  const float* x    = (const float*)d_in[0];
  const int*   ei   = (const int*)  d_in[1];
  const float* attr = (const float*)d_in[2];
  const float* W1   = (const float*)d_in[3];
  const float* b1   = (const float*)d_in[4];
  const float* W2   = (const float*)d_in[5];
  const float* b2   = (const float*)d_in[6];
  const float* W3   = (const float*)d_in[7];
  const float* b3   = (const float*)d_in[8];
  const float* W4   = (const float*)d_in[9];
  const float* b4   = (const float*)d_in[10];

  const int N = in_sizes[0] / 128;
  const int E = in_sizes[2];
  const int* src = ei;
  const int* dst = ei + E;

  char* ws = (char*)d_ws;
  size_t off = 0;
  auto wsalloc = [&](size_t bytes) {
    char* p = ws + off; off += (bytes + 255) & ~(size_t)255; return p;
  };
  float* dinv = (float*)wsalloc((size_t)N * sizeof(float));
  float* htmp = (float*)wsalloc((size_t)N * HIDF * sizeof(float));
  float* agg  = (float*)wsalloc((size_t)N * HIDF * sizeof(float));
  float* hbuf = (float*)wsalloc((size_t)N * HIDF * sizeof(float));  // h1 then h2
  float* outp = (float*)d_out;

  const int T = 256;
  const long long NH = (long long)N * HIDF;
  const long long EH = (long long)E * HIDF;
  const unsigned gN  = (unsigned)((N + T - 1) / T);
  const unsigned gE  = (unsigned)((E + T - 1) / T);
  const unsigned gNH = (unsigned)((NH + T - 1) / T);
  const unsigned gEH = (unsigned)((EH + T - 1) / T);
  const unsigned gGemm = (unsigned)(((N + 15) / 16 + 7) / 8);   // 8 waves/block
  const unsigned gMlp  = (unsigned)(((E + 15) / 16 + 7) / 8);

  // symmetric normalization coefficients
  k_deg_init <<<gN, T, 0, stream>>>(dinv, N);
  k_deg_count<<<gE, T, 0, stream>>>(dst, dinv, E);
  k_rsqrt    <<<gN, T, 0, stream>>>(dinv, N);

  // layer 1: h1 = relu(Agg(x @ W1) + b1)
  gemm_node<4><<<gGemm, T, 0, stream>>>(x, W1, htmp, N);
  k_agg_init  <<<gNH, T, 0, stream>>>(htmp, dinv, agg, NH);
  k_agg_edge  <<<gEH, T, 0, stream>>>(src, dst, dinv, htmp, agg, EH);
  k_agg_fin   <<<gNH, T, 0, stream>>>(agg, b1, hbuf, NH);

  // layer 2: h2 = relu(Agg(h1 @ W2) + b2)
  gemm_node<2><<<gGemm, T, 0, stream>>>(hbuf, W2, htmp, N);
  k_agg_init  <<<gNH, T, 0, stream>>>(htmp, dinv, agg, NH);
  k_agg_edge  <<<gEH, T, 0, stream>>>(src, dst, dinv, htmp, agg, EH);
  k_agg_fin   <<<gNH, T, 0, stream>>>(agg, b2, hbuf, NH);

  // edge MLP head
  edge_mlp<<<gMlp, T, 0, stream>>>(hbuf, src, dst, attr, W3, b3, W4, b4, outp, E);
}
